// DeformableMLPHead_14224931684774
// MI455X (gfx1250) — compile-verified, matrix operand order checked by reading, not run on hardware
//
#include <hip/hip_runtime.h>

// ---------------------------------------------------------------------------
// Deformable conv ×2 + global-avg-pool + FC, for MI455X (gfx1250, wave32).
// GEMM via v_wmma_f32_16x16x32_bf16, 2x2 register tile per wave (M=32/block)
// so each B-fragment load feeds two WMMAs and weight L2 traffic is halved.
// ---------------------------------------------------------------------------

typedef __attribute__((ext_vector_type(16))) __bf16        v16bf;
typedef __attribute__((ext_vector_type(8)))  float         v8f;
typedef __attribute__((ext_vector_type(4)))  unsigned int  v4u;

#define BATCH   16
#define CH      256
#define HH      32
#define WW      32
#define K2      9
#define KKTOT   (CH * K2)        // 2304
#define CCHUNK  64               // channels per LDS phase (4 phases)
#define KCH     (CCHUNK * K2)    // 576
#define MTILE   32               // pixels per block (one full image row)
#define NPIX    (BATCH * HH * WW)    // 16384
#define NCOORD  (MTILE * K2)     // 288
#define NCLS    1000

__device__ __forceinline__ unsigned short f2bf(float f) {
    unsigned int u = __float_as_uint(f);
    u += 0x7fffu + ((u >> 16) & 1u);   // round-to-nearest-even
    return (unsigned short)(u >> 16);
}

// --------------------------- weight fp32 -> bf16 ---------------------------
__global__ void cvt_bf16_kernel(const float* __restrict__ src,
                                unsigned short* __restrict__ dst, int n) {
    int i = blockIdx.x * 256 + threadIdx.x;
    if (i < n) dst[i] = f2bf(src[i]);
}

// --------------------------- deformable conv layer -------------------------
// One block: 32 pixels (one row: same b, same y, x=0..31) x all 256 outputs.
// 8 waves; each wave: 2 M-tiles x 2 N-tiles of 16x16 (2x2 WMMA register tile).
// mode 0: write bias+ReLU activations to `out` [B,256,32,32]
// mode 1: bias+ReLU, reduce over the 32 pixels, write block partials.
__global__ __launch_bounds__(256)
void deform_conv_kernel(const float* __restrict__ in,      // [B,256,32,32]
                        const float* __restrict__ offs,    // [B,18,32,32]
                        const unsigned short* __restrict__ wbf, // [256][2304] bf16
                        const float* __restrict__ bias,    // [256]
                        float* __restrict__ out,           // mode 0
                        float* __restrict__ partial,       // mode 1: [grid][256]
                        int mode) {
    __shared__ __align__(16) unsigned short As[MTILE * KCH];  // 36 KB im2col tile
    __shared__ int   sY0[NCOORD], sX0[NCOORD];
    __shared__ float sDY[NCOORD], sDX[NCOORD];

    const int tid = threadIdx.x;
    const int m0  = blockIdx.x * MTILE;
    const int b   = m0 >> 10;          // / (32*32)
    const int y   = (m0 >> 5) & 31;

    // ---- bilinear sampling coordinates (shared by all channels) ----
    for (int ci = tid; ci < NCOORD; ci += 256) {
        const int t = ci >> 5;         // tap 0..8
        const int p = ci & 31;         // pixel in row = x
        const float* ob = offs + (size_t)b * 18 * 1024;
        const float offy = ob[((2 * t    ) * 32 + y) * 32 + p];
        const float offx = ob[((2 * t + 1) * 32 + y) * 32 + p];
        const float sy = offy + (float)(t / 3 - 1) + (float)y;
        const float sx = offx + (float)(t % 3 - 1) + (float)p;
        const float fy = floorf(sy), fx = floorf(sx);
        sY0[ci] = (int)fy;  sX0[ci] = (int)fx;
        sDY[ci] = sy - fy;  sDX[ci] = sx - fx;
    }
    __syncthreads();

    const int lane  = tid & 31;
    const int wid   = tid >> 5;        // 0..7
    const int n0a   = wid * 32;        // this wave owns n in [n0a, n0a+32)
    const int arow  = lane & 15;       // M row (A) / N column (B)
    const int hiSel = (lane >> 4) * 8; // K sub-packing select per WMMA layout
    v8f c00 = {}, c01 = {};            // M-tile 0 x {N-tile 0, N-tile 1}
    v8f c10 = {}, c11 = {};            // M-tile 1 x {N-tile 0, N-tile 1}

    for (int cc = 0; cc < 4; ++cc) {
        // ---- phase A: gather + bilinear blend -> bf16 im2col tile ----
        const float* ib = in + (((size_t)b * CH + cc * CCHUNK) * 1024);
        for (int i = tid; i < MTILE * KCH; i += 256) {
            const int c  = i / NCOORD;     // 0..63
            const int r  = i - c * NCOORD;
            const int t  = r >> 5;
            const int p  = r & 31;
            const int ci = (t << 5) + p;
            const int y0 = sY0[ci], x1 = sX0[ci];
            const float dy = sDY[ci], dx = sDX[ci];
            const float* img = ib + (size_t)c * 1024;
            float v00 = 0.f, v01 = 0.f, v10 = 0.f, v11 = 0.f;
            const bool yi0 = (unsigned)y0 < 32u, yi1 = (unsigned)(y0 + 1) < 32u;
            const bool xi0 = (unsigned)x1 < 32u, xi1 = (unsigned)(x1 + 1) < 32u;
            if (yi0 & xi0) v00 = img[y0 * 32 + x1];
            if (yi0 & xi1) v01 = img[y0 * 32 + x1 + 1];
            if (yi1 & xi0) v10 = img[(y0 + 1) * 32 + x1];
            if (yi1 & xi1) v11 = img[(y0 + 1) * 32 + x1 + 1];
            const float val = v00 * (1.f - dy) * (1.f - dx) + v01 * (1.f - dy) * dx
                            + v10 * dy * (1.f - dx)         + v11 * dy * dx;
            As[p * KCH + c * 9 + t] = f2bf(val);
        }
        __syncthreads();

        // ---- phase B: WMMA over this K chunk (2x2 register tile) ----
        const int kbase = cc * KCH;
        for (int k0 = 0; k0 < KCH; k0 += 32) {
            union { v16bf v; v4u u[2]; } fa0, fa1, fb0, fb1;
            const unsigned short* ap0 = &As[(arow     ) * KCH + k0 + hiSel];
            const unsigned short* ap1 = &As[(16 + arow) * KCH + k0 + hiSel];
            fa0.u[0] = *(const v4u*)(ap0);  fa0.u[1] = *(const v4u*)(ap0 + 16);
            fa1.u[0] = *(const v4u*)(ap1);  fa1.u[1] = *(const v4u*)(ap1 + 16);
            const int kg = kbase + k0 + hiSel;
            const unsigned short* bp0 = wbf + (size_t)(n0a      + arow) * KKTOT + kg;
            const unsigned short* bp1 = wbf + (size_t)(n0a + 16 + arow) * KKTOT + kg;
            fb0.u[0] = *(const v4u*)(bp0);  fb0.u[1] = *(const v4u*)(bp0 + 16);
            fb1.u[0] = *(const v4u*)(bp1);  fb1.u[1] = *(const v4u*)(bp1 + 16);
            c00 = __builtin_amdgcn_wmma_f32_16x16x32_bf16(false, fa0.v, false, fb0.v,
                                                          (short)0, c00, false, false);
            c10 = __builtin_amdgcn_wmma_f32_16x16x32_bf16(false, fa1.v, false, fb0.v,
                                                          (short)0, c10, false, false);
            c01 = __builtin_amdgcn_wmma_f32_16x16x32_bf16(false, fa0.v, false, fb1.v,
                                                          (short)0, c01, false, false);
            c11 = __builtin_amdgcn_wmma_f32_16x16x32_bf16(false, fa1.v, false, fb1.v,
                                                          (short)0, c11, false, false);
        }
        __syncthreads();
    }

    // C/D layout: lane L holds n = L%16; VGPR r holds m = r + 8*(L/16)
    const int nlo  = lane & 15;
    const int mofs = (lane >> 4) * 8;
    const float bia0 = bias[n0a + nlo];
    const float bia1 = bias[n0a + 16 + nlo];

    if (mode == 0) {
        float* ob0 = out + (((size_t)b * CH + n0a + nlo     ) * 32 + y) * 32;
        float* ob1 = out + (((size_t)b * CH + n0a + 16 + nlo) * 32 + y) * 32;
#pragma unroll
        for (int r = 0; r < 8; ++r) {
            const int m = mofs + r;                 // pixel x for M-tile 0
            float v;
            v = c00[r] + bia0; ob0[m]      = v > 0.f ? v : 0.f;
            v = c10[r] + bia0; ob0[m + 16] = v > 0.f ? v : 0.f;
            v = c01[r] + bia1; ob1[m]      = v > 0.f ? v : 0.f;
            v = c11[r] + bia1; ob1[m + 16] = v > 0.f ? v : 0.f;
        }
    } else {
        float s0 = 0.f, s1 = 0.f;
#pragma unroll
        for (int r = 0; r < 8; ++r) {
            float v;
            v = c00[r] + bia0; s0 += v > 0.f ? v : 0.f;
            v = c10[r] + bia0; s0 += v > 0.f ? v : 0.f;
            v = c01[r] + bia1; s1 += v > 0.f ? v : 0.f;
            v = c11[r] + bia1; s1 += v > 0.f ? v : 0.f;
        }
        s0 += __shfl_xor(s0, 16, 32);
        s1 += __shfl_xor(s1, 16, 32);
        if (lane < 16) {
            float* pr = partial + (size_t)blockIdx.x * 256;
            pr[n0a + nlo]      = s0;
            pr[n0a + 16 + nlo] = s1;
        }
    }
}

// ------------------- deterministic pooled reduction ------------------------
__global__ void pool_reduce_kernel(const float* __restrict__ partial,
                                   float* __restrict__ pooled) {
    const int i = blockIdx.x * 256 + threadIdx.x;   // 16*256 = 4096
    if (i >= BATCH * CH) return;
    const int b = i >> 8, n = i & 255;
    const float* p = partial + (size_t)(b * 32) * 256 + n;   // 32 blocks per batch
    float s = 0.f;
#pragma unroll 4
    for (int q = 0; q < 32; ++q) s += p[q * 256];
    pooled[i] = s;
}

// ------------------------------- FC head -----------------------------------
__global__ void fc_kernel(const float* __restrict__ pooled,  // [16][256] sums
                          const float* __restrict__ fcw,     // [1000][256]
                          const float* __restrict__ fcb,     // [1000]
                          float* __restrict__ outp) {        // [16][1000]
    const int i = blockIdx.x * 256 + threadIdx.x;
    if (i >= BATCH * NCLS) return;
    const int b = i / NCLS, j = i - b * NCLS;
    const float* pv = pooled + b * 256;
    const float* wv = fcw + (size_t)j * 256;
    float acc = 0.f;
#pragma unroll 4
    for (int o = 0; o < 256; ++o) acc += pv[o] * wv[o];
    outp[i] = acc * (1.f / (HH * WW)) + fcb[j];
}

// ---------------------------------------------------------------------------
extern "C" void kernel_launch(void* const* d_in, const int* in_sizes, int n_in,
                              void* d_out, int out_size, void* d_ws, size_t ws_size,
                              hipStream_t stream) {
    const float* x    = (const float*)d_in[0];
    const float* offs = (const float*)d_in[1];
    const float* w1   = (const float*)d_in[2];
    const float* b1   = (const float*)d_in[3];
    const float* w2   = (const float*)d_in[4];
    const float* b2   = (const float*)d_in[5];
    const float* fcw  = (const float*)d_in[6];
    const float* fcb  = (const float*)d_in[7];
    float* outp = (float*)d_out;

    // workspace carve-up (byte offsets, 16B-aligned)
    char* ws = (char*)d_ws;
    const size_t nW = (size_t)CH * KKTOT;                 // 589824 elems / weight
    unsigned short* wbf1 = (unsigned short*)(ws);                         // 1.18 MB
    unsigned short* wbf2 = (unsigned short*)(ws + 1179648);               // 1.18 MB
    float* h1      = (float*)(ws + 2359296);                              // 16.8 MB
    float* partial = (float*)(ws + 2359296 + 16777216);                   // 512 KB
    float* pooled  = (float*)(ws + 2359296 + 16777216 + 524288);          // 16 KB

    // 1) weights to bf16
    {
        const int nblk = (int)((nW + 255) / 256);
        cvt_bf16_kernel<<<nblk, 256, 0, stream>>>(w1, wbf1, (int)nW);
        cvt_bf16_kernel<<<nblk, 256, 0, stream>>>(w2, wbf2, (int)nW);
    }

    // 2) layer 1: deform-conv + ReLU -> h1
    deform_conv_kernel<<<NPIX / MTILE, 256, 0, stream>>>(x, offs, wbf1, b1,
                                                         h1, nullptr, 0);

    // 3) layer 2: deform-conv + ReLU -> per-block pooled partials
    deform_conv_kernel<<<NPIX / MTILE, 256, 0, stream>>>(h1, offs, wbf2, b2,
                                                         nullptr, partial, 1);

    // 4) deterministic pooled reduction
    pool_reduce_kernel<<<(BATCH * CH + 255) / 256, 256, 0, stream>>>(partial, pooled);

    // 5) FC head
    fc_kernel<<<(BATCH * NCLS + 255) / 256, 256, 0, stream>>>(pooled, fcw, fcb, outp);
}
